// SwinTransformerBlock_86311662780635
// MI455X (gfx1250) — matrix operand order repaired and checked by reading, not compiled, not run on hardware
//
#include <hip/hip_runtime.h>
#include <hip/hip_bf16.h>

typedef __attribute__((ext_vector_type(16))) _Float16 v16h;
typedef __attribute__((ext_vector_type(8)))  float    v8f;
typedef __attribute__((ext_vector_type(4)))  unsigned int u32x4;
typedef __attribute__((ext_vector_type(4)))  int i32x4;
typedef __attribute__((ext_vector_type(8)))  int i32x8;

union FragU { v16h v; u32x4 q[2]; };

// ---------------- WMMA fragment helpers (CDNA5 16x16x32 f16 layouts) -------------
// A fragment: 16x32 tile, row-major source with stride (halfs).
// lane: m = mbase + (lane&15); hi = lane>>4; elems 0..7 -> K = kbase+8*hi+e,
// elems 8..15 -> K = kbase+16+8*hi+(e-8). Both chunks 16B-aligned contiguous.
__device__ __forceinline__ v16h frag_a(const _Float16* p, int stride, int mbase,
                                       int kbase, int lane) {
    int m = mbase + (lane & 15);
    int hi = lane >> 4;
    const _Float16* r = p + m * stride + kbase + 8 * hi;
    FragU f;
    f.q[0] = *(const u32x4*)(r);
    f.q[1] = *(const u32x4*)(r + 16);
    return f.v;
}

// B fragment where B[k][n] = W[n][k], W row-major with leading dim ldw (halfs).
// lane: n = nbase + (lane&15); hi = lane>>4; elems 0..7 -> K = kbase+8*hi+e,
// elems 8..15 -> K = kbase+16+8*hi+(e-8).
__device__ __forceinline__ v16h frag_b_wt(const _Float16* w, int ldw, int nbase,
                                          int kbase, int lane) {
    int n = nbase + (lane & 15);
    int hi = lane >> 4;
    const _Float16* r = w + n * ldw + kbase + 8 * hi;
    FragU f;
    f.q[0] = *(const u32x4*)(r);
    f.q[1] = *(const u32x4*)(r + 16);
    return f.v;
}

__device__ __forceinline__ v8f wmma_f16(v16h a, v16h b, v8f c) {
    return __builtin_amdgcn_wmma_f32_16x16x32_f16(false, a, false, b, (short)0, c,
                                                  false, false);
}

__device__ __forceinline__ float wave_sum(float v) {
#pragma unroll
    for (int m = 16; m >= 1; m >>= 1) v += __shfl_xor(v, m, 32);
    return v;
}
__device__ __forceinline__ float wave_max(float v) {
#pragma unroll
    for (int m = 16; m >= 1; m >>= 1) v = fmaxf(v, __shfl_xor(v, m, 32));
    return v;
}

// Geometry constants
#define BATCH 2
#define DD 32
#define HH 64
#define WW 64
#define CC 96
#define NWIN 4096         // 2 * 8 * 16 * 16
#define NTOK 64           // 4*4*4 tokens per window
#define MLPH 384

#if defined(__has_builtin)
#if __has_builtin(__builtin_amdgcn_tensor_load_to_lds) && \
    __has_builtin(__builtin_amdgcn_s_wait_tensorcnt)
#define HAVE_TDM 1
#endif
#endif

// ---------------- K0: fp32 -> f16 weight cast -----------------------------------
__global__ void k_cvt(const float* __restrict__ s, _Float16* __restrict__ d, int n) {
    int i = blockIdx.x * 256 + threadIdx.x;
    if (i < n) d[i] = (_Float16)s[i];
}

// ---------------- K1: LN1 + cyclic shift + window partition (one wave/token) ----
__global__ __launch_bounds__(256) void k_ln1_part(const float* __restrict__ x,
                                                  const float* __restrict__ lw,
                                                  const float* __restrict__ lb,
                                                  _Float16* __restrict__ xw) {
    int wave = threadIdx.x >> 5, lane = threadIdx.x & 31;
    int token = blockIdx.x * 8 + wave;           // shifted-space flat token
    int wc = token & 63;  int t = token >> 6;
    int hc = t & 63;      t >>= 6;
    int dc = t & 31;      int b = t >> 5;
    int sd = (dc + 2) & 31, sh = (hc + 2) & 63, sw = (wc + 2) & 63;
    const float* px = x + ((((b * DD + sd) * HH + sh) * WW + sw)) * (long)CC + 3 * lane;
    float v0 = px[0], v1 = px[1], v2 = px[2];
    float mu = wave_sum(v0 + v1 + v2) * (1.f / 96.f);
    float d0 = v0 - mu, d1 = v1 - mu, d2 = v2 - mu;
    float var = wave_sum(d0 * d0 + d1 * d1 + d2 * d2) * (1.f / 96.f);
    float rs = rsqrtf(var + 1e-5f);
    int wd = dc >> 2, wh = hc >> 2, wwi = wc >> 2;
    int win = ((b * 8 + wd) * 16 + wh) * 16 + wwi;
    int n = ((dc & 3) << 4) | ((hc & 3) << 2) | (wc & 3);
    _Float16* pd = xw + ((long)win * NTOK + n) * CC + 3 * lane;
    int c = 3 * lane;
    pd[0] = (_Float16)(d0 * rs * lw[c]     + lb[c]);
    pd[1] = (_Float16)(d1 * rs * lw[c + 1] + lb[c + 1]);
    pd[2] = (_Float16)(d2 * rs * lw[c + 2] + lb[c + 2]);
}

// ---------------- K2: per-window fused attention ---------------------------------
// LDS: X(12K) Q(12K) K(12K) Vt(12K, transposed 96x64) O(12K) S(16K f32) P(8K f16)
__global__ __launch_bounds__(256) void k_attn(const _Float16* __restrict__ xw,
                                              const _Float16* __restrict__ qkv_w,
                                              const float* __restrict__ qkv_b,
                                              const _Float16* __restrict__ proj_w,
                                              const float* __restrict__ proj_b,
                                              const float* __restrict__ x,
                                              float* __restrict__ out) {
    extern __shared__ char smem[];
    _Float16* Xs = (_Float16*)smem;
    _Float16* Qs = Xs + NTOK * CC;
    _Float16* Ks = Qs + NTOK * CC;
    _Float16* Vt = Ks + NTOK * CC;      // [96][64] transposed
    _Float16* Os = Vt + NTOK * CC;
    float*    Ss = (float*)(Os + NTOK * CC);
    _Float16* Ps = (_Float16*)(Ss + NTOK * NTOK);

    int tid = threadIdx.x, wave = tid >> 5, lane = tid & 31;
    int win = blockIdx.x;
    int mt = wave & 3, nth = wave >> 2;     // 4 m-tiles x 2 wave-columns

    // ---- Stage 0: window activations global -> LDS ----
#ifdef HAVE_TDM
    if (wave == 0) {
        // Tensor Data Mover: 2D tile, 48 dwords x 64 rows, contiguous.
        unsigned lds_off = (unsigned)(unsigned long long)Xs;
        unsigned long long ga =
            (unsigned long long)(xw + (long)win * (NTOK * CC));
        u32x4 g0;
        g0[0] = 1u;                                  // count=1, user descriptor
        g0[1] = lds_off;                             // lds_addr (bytes)
        g0[2] = (unsigned)ga;                        // global_addr[31:0]
        g0[3] = (unsigned)((ga >> 32) & 0x1FFFFFFu)  // global_addr[56:32]
                | (2u << 30);                        // type=2 (image)
        i32x8 g1;
        g1[0] = (2 << 16);        // wg_mask=0, data_size=2 -> 4 bytes
        g1[1] = (48 << 16);       // tensor_dim0 = 48 dwords (bits 79:48, low half)
        g1[2] = (64 << 16);       // tensor_dim0 hi=0 | tensor_dim1 = 64
        g1[3] = (48 << 16);       // tensor_dim1 hi=0 | tile_dim0 = 48
        g1[4] = 64;               // tile_dim1 = 64, tile_dim2 = 0
        g1[5] = 48;               // tensor_dim0_stride = 48 dwords
        g1[6] = 0;                // stride hi, tensor_dim1_stride lo
        g1[7] = 0;
        i32x4 gz = {0, 0, 0, 0};
#if __clang_major__ >= 23
        i32x8 gz8 = {0, 0, 0, 0, 0, 0, 0, 0};
        __builtin_amdgcn_tensor_load_to_lds(g0, g1, gz, gz, gz8, 0);
#else
        __builtin_amdgcn_tensor_load_to_lds(g0, g1, gz, gz, 0);
#endif
        __builtin_amdgcn_s_wait_tensorcnt(0);
    }
#else
    {
        const u32x4* src = (const u32x4*)(xw + (long)win * (NTOK * CC));
        u32x4* dst = (u32x4*)Xs;
#pragma unroll
        for (int i = 0; i < 3; ++i) dst[tid + 256 * i] = src[tid + 256 * i];
    }
#endif
    __syncthreads();

    // ---- QKV GEMM: 64x288, K=96. A-frags pinned per wave, stream 9 B-tiles ----
    {
        v16h a0 = frag_a(Xs, CC, mt * 16, 0, lane);
        v16h a1 = frag_a(Xs, CC, mt * 16, 32, lane);
        v16h a2 = frag_a(Xs, CC, mt * 16, 64, lane);
        for (int i = 0; i < 9; ++i) {
            int nt = nth + 2 * i;
            v8f acc = {};
            acc = wmma_f16(a0, frag_b_wt(qkv_w, CC, nt * 16, 0, lane), acc);
            acc = wmma_f16(a1, frag_b_wt(qkv_w, CC, nt * 16, 32, lane), acc);
            acc = wmma_f16(a2, frag_b_wt(qkv_w, CC, nt * 16, 64, lane), acc);
            int n = nt * 16 + (lane & 15);
            float bias = qkv_b[n];
            int hi = lane >> 4;
            if (n < 192) {                       // Q / K row-major
                _Float16* dstm = (n < 96) ? Qs : Ks;
                int nn = (n < 96) ? n : n - 96;
#pragma unroll
                for (int j = 0; j < 8; ++j)
                    dstm[(mt * 16 + 8 * hi + j) * CC + nn] = (_Float16)(acc[j] + bias);
            } else {                             // V transposed: one b128 store
                int nn = n - 192;
                union { u32x4 q; _Float16 h[8]; } pk;
#pragma unroll
                for (int j = 0; j < 8; ++j) pk.h[j] = (_Float16)(acc[j] + bias);
                *(u32x4*)(Vt + nn * 64 + mt * 16 + 8 * hi) = pk.q;
            }
        }
    }
    __syncthreads();

    const float scale = 0.17677669529663687f;    // 1/sqrt(32)
    for (int head = 0; head < 3; ++head) {
        int hoff = head * 32;
        // S = (Q @ K^T) * scale : 16 tiles, K = 32; A-frag reused across 2 tiles
        {
            v16h a = frag_a(Qs + hoff, CC, mt * 16, 0, lane);
#pragma unroll
            for (int i = 0; i < 2; ++i) {
                int nt = nth * 2 + i;
                v8f acc = {};
                acc = wmma_f16(a, frag_b_wt(Ks + hoff, CC, nt * 16, 0, lane), acc);
                int n = nt * 16 + (lane & 15), hi = lane >> 4;
#pragma unroll
                for (int j = 0; j < 8; ++j)
                    Ss[(mt * 16 + 8 * hi + j) * NTOK + n] = acc[j] * scale;
            }
        }
        __syncthreads();
        // row softmax: wave handles 8 rows, contiguous 2 cols per lane
#pragma unroll
        for (int i = 0; i < 8; ++i) {
            int r = wave * 8 + i, c = 2 * lane;
            float a0 = Ss[r * NTOK + c], a1 = Ss[r * NTOK + c + 1];
            float mx = wave_max(fmaxf(a0, a1));
            float e0 = __expf(a0 - mx), e1 = __expf(a1 - mx);
            float inv = 1.f / wave_sum(e0 + e1);
            union { unsigned u; _Float16 h[2]; } pk;
            pk.h[0] = (_Float16)(e0 * inv);
            pk.h[1] = (_Float16)(e1 * inv);
            *(unsigned*)(Ps + r * NTOK + c) = pk.u;
        }
        __syncthreads();
        // O_h = P @ V_h : 8 tiles (one per wave), K = 64, Vt contiguous B-frags
        {
            v8f acc = {};
            v16h a0 = frag_a(Ps, NTOK, mt * 16, 0, lane);
            v16h a1 = frag_a(Ps, NTOK, mt * 16, 32, lane);
            acc = wmma_f16(a0, frag_b_wt(Vt + hoff * 64, 64, nth * 16, 0, lane), acc);
            acc = wmma_f16(a1, frag_b_wt(Vt + hoff * 64, 64, nth * 16, 32, lane), acc);
            int n = nth * 16 + (lane & 15), hi = lane >> 4;
#pragma unroll
            for (int j = 0; j < 8; ++j)
                Os[(mt * 16 + 8 * hi + j) * CC + hoff + n] = (_Float16)acc[j];
        }
        __syncthreads();
    }

    // ---- proj GEMM + roll-back + residual epilogue ----
    int b   = win >> 11;
    int rem = win & 2047;
    int wd = rem >> 8, wh = (rem >> 4) & 15, wwi = rem & 15;
    {
        v16h a0 = frag_a(Os, CC, mt * 16, 0, lane);
        v16h a1 = frag_a(Os, CC, mt * 16, 32, lane);
        v16h a2 = frag_a(Os, CC, mt * 16, 64, lane);
#pragma unroll
        for (int i = 0; i < 3; ++i) {
            int nt = nth * 3 + i;
            v8f acc = {};
            acc = wmma_f16(a0, frag_b_wt(proj_w, CC, nt * 16, 0, lane), acc);
            acc = wmma_f16(a1, frag_b_wt(proj_w, CC, nt * 16, 32, lane), acc);
            acc = wmma_f16(a2, frag_b_wt(proj_w, CC, nt * 16, 64, lane), acc);
            int n = nt * 16 + (lane & 15), hi = lane >> 4;
            float bias = proj_b[n];
#pragma unroll
            for (int j = 0; j < 8; ++j) {
                int m = mt * 16 + 8 * hi + j;
                int dl = m >> 4, hl = (m >> 2) & 3, wl = m & 3;
                int dd = (wd * 4 + dl + 2) & 31;
                int hh = (wh * 4 + hl + 2) & 63;
                int wo = (wwi * 4 + wl + 2) & 63;
                long idx = ((((long)b * DD + dd) * HH + hh) * WW + wo) * CC + n;
                out[idx] = x[idx] + acc[j] + bias;
            }
        }
    }
}

// ---------------- K3: fused LN2 + FC1 + GELU + FC2 + residual (in-place) --------
// LDS: A(12K f16) + H(48K f16) = 60KB
__global__ __launch_bounds__(256) void k_mlp(float* __restrict__ io,
                                             const float* __restrict__ lw,
                                             const float* __restrict__ lb,
                                             const _Float16* __restrict__ fc1_w,
                                             const float* __restrict__ fc1_b,
                                             const _Float16* __restrict__ fc2_w,
                                             const float* __restrict__ fc2_b) {
    extern __shared__ char smem[];
    _Float16* As = (_Float16*)smem;          // 64 x 96
    _Float16* Hs = As + 64 * CC;             // 64 x 384
    int tid = threadIdx.x, wave = tid >> 5, lane = tid & 31;
    int mt = wave & 3, nth = wave >> 2;
    long tok0 = (long)blockIdx.x * 64;

    // LN2: one wave per token (8 tokens/wave), contiguous 3 channels per lane
#pragma unroll
    for (int i = 0; i < 8; ++i) {
        int r = wave * 8 + i, c = 3 * lane;
        const float* px = io + (tok0 + r) * CC + c;
        float v0 = px[0], v1 = px[1], v2 = px[2];
        float mu = wave_sum(v0 + v1 + v2) * (1.f / 96.f);
        float d0 = v0 - mu, d1 = v1 - mu, d2 = v2 - mu;
        float var = wave_sum(d0 * d0 + d1 * d1 + d2 * d2) * (1.f / 96.f);
        float rs = rsqrtf(var + 1e-5f);
        _Float16* pd = As + r * CC + c;
        pd[0] = (_Float16)(d0 * rs * lw[c]     + lb[c]);
        pd[1] = (_Float16)(d1 * rs * lw[c + 1] + lb[c + 1]);
        pd[2] = (_Float16)(d2 * rs * lw[c + 2] + lb[c + 2]);
    }
    __syncthreads();

    // FC1 + exact GELU: 64x384, K=96; A-frags pinned, stream 12 B-tiles
    {
        v16h a0 = frag_a(As, CC, mt * 16, 0, lane);
        v16h a1 = frag_a(As, CC, mt * 16, 32, lane);
        v16h a2 = frag_a(As, CC, mt * 16, 64, lane);
        for (int i = 0; i < 12; ++i) {
            int nt = nth + 2 * i;
            v8f acc = {};
            acc = wmma_f16(a0, frag_b_wt(fc1_w, CC, nt * 16, 0, lane), acc);
            acc = wmma_f16(a1, frag_b_wt(fc1_w, CC, nt * 16, 32, lane), acc);
            acc = wmma_f16(a2, frag_b_wt(fc1_w, CC, nt * 16, 64, lane), acc);
            int n = nt * 16 + (lane & 15), hi = lane >> 4;
            float bias = fc1_b[n];
#pragma unroll
            for (int j = 0; j < 8; ++j) {
                float v = acc[j] + bias;
                float g = 0.5f * v * (1.f + erff(v * 0.70710678118654752f));
                Hs[(mt * 16 + 8 * hi + j) * MLPH + n] = (_Float16)g;
            }
        }
    }
    __syncthreads();

    // FC2 + residual: 64x96, K=384; hold all 12 A-frags in registers
    {
        v16h af[12];
#pragma unroll
        for (int k = 0; k < 12; ++k)
            af[k] = frag_a(Hs, MLPH, mt * 16, k * 32, lane);
#pragma unroll
        for (int i = 0; i < 3; ++i) {
            int nt = nth * 3 + i;
            v8f acc = {};
#pragma unroll
            for (int k = 0; k < 12; ++k)
                acc = wmma_f16(af[k], frag_b_wt(fc2_w, MLPH, nt * 16, k * 32, lane),
                               acc);
            int n = nt * 16 + (lane & 15), hi = lane >> 4;
            float bias = fc2_b[n];
#pragma unroll
            for (int j = 0; j < 8; ++j) {
                long idx = (tok0 + mt * 16 + 8 * hi + j) * CC + n;
                io[idx] = io[idx] + acc[j] + bias;
            }
        }
    }
}

extern "C" void kernel_launch(void* const* d_in, const int* in_sizes, int n_in,
                              void* d_out, int out_size, void* d_ws, size_t ws_size,
                              hipStream_t stream) {
    (void)in_sizes; (void)n_in; (void)out_size; (void)ws_size;
    const float* x      = (const float*)d_in[0];
    const float* ln1_w  = (const float*)d_in[1];
    const float* ln1_b  = (const float*)d_in[2];
    const float* qkv_w  = (const float*)d_in[3];
    const float* qkv_b  = (const float*)d_in[4];
    const float* proj_w = (const float*)d_in[5];
    const float* proj_b = (const float*)d_in[6];
    const float* ln2_w  = (const float*)d_in[7];
    const float* ln2_b  = (const float*)d_in[8];
    const float* fc1_w  = (const float*)d_in[9];
    const float* fc1_b  = (const float*)d_in[10];
    const float* fc2_w  = (const float*)d_in[11];
    const float* fc2_b  = (const float*)d_in[12];
    float* out = (float*)d_out;

    char* ws = (char*)d_ws;
    size_t off = 0;
    _Float16* xw = (_Float16*)(ws + off); off += (size_t)NWIN * NTOK * CC * 2;   // 48 MB
    _Float16* qkv16  = (_Float16*)(ws + off); off += (size_t)288 * 96 * 2;
    _Float16* proj16 = (_Float16*)(ws + off); off += (size_t)96 * 96 * 2;
    _Float16* fc1_16 = (_Float16*)(ws + off); off += (size_t)384 * 96 * 2;
    _Float16* fc2_16 = (_Float16*)(ws + off); off += (size_t)96 * 384 * 2;

    k_cvt<<<(288 * 96 + 255) / 256, 256, 0, stream>>>(qkv_w, qkv16, 288 * 96);
    k_cvt<<<(96 * 96 + 255) / 256, 256, 0, stream>>>(proj_w, proj16, 96 * 96);
    k_cvt<<<(384 * 96 + 255) / 256, 256, 0, stream>>>(fc1_w, fc1_16, 384 * 96);
    k_cvt<<<(96 * 384 + 255) / 256, 256, 0, stream>>>(fc2_w, fc2_16, 96 * 384);

    k_ln1_part<<<BATCH * DD * HH * WW / 8, 256, 0, stream>>>(x, ln1_w, ln1_b, xw);

    size_t attn_lds = (size_t)(5 * NTOK * CC * 2 + NTOK * NTOK * 4 + NTOK * NTOK * 2);
    k_attn<<<NWIN, 256, attn_lds, stream>>>(xw, qkv16, qkv_b, proj16, proj_b, x, out);

    size_t mlp_lds = (size_t)(64 * CC * 2 + 64 * MLPH * 2);
    k_mlp<<<BATCH * DD * HH * WW / 64, 256, mlp_lds, stream>>>(
        out, ln2_w, ln2_b, fc1_16, fc1_b, fc2_16, fc2_b);
}
